// MultiHeadAttention_9268539424885
// MI455X (gfx1250) — matrix operand order repaired and checked by reading, not compiled, not run on hardware
//
#include <hip/hip_runtime.h>
#include <hip/hip_bf16.h>
#include <math.h>

typedef __attribute__((ext_vector_type(16))) _Float16 v16h;
typedef __attribute__((ext_vector_type(8)))  _Float16 v8h;
typedef __attribute__((ext_vector_type(4)))  _Float16 v4h;
typedef __attribute__((ext_vector_type(8)))  float    v8f;

#define DIMC 1024
#define NH   16
#define HD   64
#define BB   2
#define SS   2048

// LDS row pitches (halves) chosen for conflict-free ds_load_b128:
// K pitch 72 -> 36 dwords/row, 36*i mod 64 distinct for i=0..15
// V/P pitch 40 -> 20 dwords/row, 20*i mod 64 distinct for i=0..15
#define KPITCH 72
#define VPITCH 40
#define PPITCH 40

// ---------------------------------------------------------------------------
// WMMA helper: D = A(16x32 f16) * B(32x16 f16) + C(16x16 f32)
// ---------------------------------------------------------------------------
__device__ __forceinline__ v8f wmma_f16(v16h a, v16h b, v8f c) {
  return __builtin_amdgcn_wmma_f32_16x16x32_f16(
      /*neg_a=*/false, a, /*neg_b=*/false, b,
      /*c_mod=*/(short)0, c, /*reuse_a=*/false, /*reuse_b=*/false);
}

// A-operand (16-bit, 16x32) per-lane gather: row m = lane%16, half = lane/16.
// elements 0..15 = K[half*16 .. +15] (contiguous)
__device__ __forceinline__ v16h load_a16(const _Float16* p, int half) {
  union { v16h v; v8h h[2]; } u;
  u.h[0] = *(const v8h*)(p + half * 16);
  u.h[1] = *(const v8h*)(p + half * 16 + 8);
  return u.v;
}

// B-operand: 16 contiguous halves (32B) as two 16B loads (16B alignment only)
__device__ __forceinline__ v16h load_b16(const _Float16* p) {
  union { v16h v; v8h h[2]; } u;
  u.h[0] = *(const v8h*)(p);
  u.h[1] = *(const v8h*)(p + 8);
  return u.v;
}

// ---------------------------------------------------------------------------
// Kernel 1: f32 -> f16 pack (activations), 4 elems/thread
// ---------------------------------------------------------------------------
__global__ void cvt_f32_f16_kernel(const float* __restrict__ src,
                                   _Float16* __restrict__ dst) {
  size_t i = ((size_t)blockIdx.x * blockDim.x + threadIdx.x) * 4;
  float4 v = *(const float4*)(src + i);
  v4h o;
  o[0] = (_Float16)v.x; o[1] = (_Float16)v.y;
  o[2] = (_Float16)v.z; o[3] = (_Float16)v.w;
  *(v4h*)(dst + i) = o;
}

// ---------------------------------------------------------------------------
// Kernel 2: W [H, D, hd] f32 -> Wt [H, hd, D] f16 (B-operand: K contiguous)
// ---------------------------------------------------------------------------
__global__ void pack_w_kernel(const float* __restrict__ W,
                              _Float16* __restrict__ Wt) {
  int i = blockIdx.x * blockDim.x + threadIdx.x;   // i = h*65536 + n*1024 + d
  int h = i >> 16;
  int n = (i >> 10) & 63;
  int d = i & 1023;
  Wt[i] = (_Float16)W[(h << 16) | (d << 6) | n];
}

// ---------------------------------------------------------------------------
// Kernel 3: projection. One wave -> 16 rows x 64 cols for one (b,h).
// mode 0: q (scale 1/8, row-major [B,H,S,hd])
// mode 1: k (row-major [B,H,S,hd])
// mode 2: v (transposed [B,H,hd,S])
// ---------------------------------------------------------------------------
__global__ void proj_kernel(const _Float16* __restrict__ X,
                            const _Float16* __restrict__ Wt,
                            const float* __restrict__ bias,
                            _Float16* __restrict__ out, int mode) {
  int lane = threadIdx.x & 31;
  int wave = threadIdx.x >> 5;
  int n16 = lane & 15;
  int half = lane >> 4;
  int h = blockIdx.y;
  int b = blockIdx.z;
  int row0 = (blockIdx.x * 4 + wave) * 16;

  const _Float16* xrow = X + ((size_t)b * SS + row0 + n16) * DIMC;  // A row m=n16
  const _Float16* wt = Wt + (size_t)h * (HD * DIMC);

  v8f acc[4] = {};
#pragma unroll 4
  for (int k0 = 0; k0 < DIMC; k0 += 32) {
    v16h a = load_a16(xrow + k0, half);
#pragma unroll
    for (int nt = 0; nt < 4; ++nt) {
      v16h bb = *(const v16h*)(wt + (size_t)(nt * 16 + n16) * DIMC + k0 + half * 16);
      acc[nt] = wmma_f16(a, bb, acc[nt]);
    }
  }

  if (mode == 2) {
    // vT[b,h,n,row]; C-layout rows r+8*half are contiguous -> v8h store
#pragma unroll
    for (int nt = 0; nt < 4; ++nt) {
      int n = nt * 16 + n16;
      float bv = bias[h * HD + n];
      v8h pk;
#pragma unroll
      for (int r = 0; r < 8; ++r) pk[r] = (_Float16)(acc[nt][r] + bv);
      _Float16* dst = out + ((size_t)(b * NH + h) * HD + n) * SS + row0 + half * 8;
      *(v8h*)dst = pk;
    }
  } else {
    float scale = (mode == 0) ? 0.125f : 1.0f;  // fold 1/sqrt(hd) into q
#pragma unroll
    for (int nt = 0; nt < 4; ++nt) {
      int n = nt * 16 + n16;
      float bv = bias[h * HD + n];
#pragma unroll
      for (int r = 0; r < 8; ++r) {
        int row = row0 + r + 8 * half;
        out[((size_t)(b * NH + h) * SS + row) * HD + n] =
            (_Float16)((acc[nt][r] + bv) * scale);
      }
    }
  }
}

// ---------------------------------------------------------------------------
// Kernel 4: flash attention. 4 waves/block, 16 query rows/wave, one (b,h).
// K (32x64) and V^T (64x32) tiles staged in LDS once per block (4x reuse),
// register double-buffered so next tile's global loads overlap compute.
// ---------------------------------------------------------------------------
__global__ void attn_kernel(const _Float16* __restrict__ qp,
                            const _Float16* __restrict__ kp,
                            const _Float16* __restrict__ vT,
                            float* __restrict__ att) {
  __shared__ _Float16 ldsK[32 * KPITCH];      // 4.5 KB
  __shared__ _Float16 ldsV[64 * VPITCH];      // 5.0 KB
  __shared__ _Float16 ldsP[4][16 * PPITCH];   // 5.0 KB (per-wave P bounce)

  const int tid = threadIdx.x;
  const int lane = tid & 31;
  const int wave = tid >> 5;
  const int n16 = lane & 15;
  const int half = lane >> 4;
  const int h = blockIdx.y;
  const int b = blockIdx.z;
  const int row0 = (blockIdx.x * 4 + wave) * 16;
  const size_t bh = (size_t)(b * NH + h);

  const _Float16* qrow = qp + (bh * SS + row0 + n16) * HD;
  v16h a0 = load_a16(qrow, half);        // q, K = hd 0..31
  v16h a1 = load_a16(qrow + 32, half);   // q, K = hd 32..63
  const _Float16* kb = kp + bh * (SS * HD);
  const _Float16* vb = vT + bh * (HD * SS);
  _Float16* P = ldsP[wave];

  // cooperative staging: 128 threads x 32B each per tile
  const int kr = tid >> 2, kc = tid & 3;   // K tile: row 0..31, 16-half chunk 0..3
  const int vr = tid >> 1, vc = tid & 1;   // V tile: row 0..63, 16-half chunk 0..1

  union { v16h v; v8h h2[2]; } kreg, vreg;
  {
    const _Float16* kpp = kb + (size_t)kr * HD + kc * 16;
    kreg.h2[0] = *(const v8h*)kpp;
    kreg.h2[1] = *(const v8h*)(kpp + 8);
    const _Float16* vpp = vb + (size_t)vr * SS + vc * 16;
    vreg.h2[0] = *(const v8h*)vpp;
    vreg.h2[1] = *(const v8h*)(vpp + 8);
  }

  v8f O[4] = {};
  float m[8], l[8];
#pragma unroll
  for (int r = 0; r < 8; ++r) { m[r] = -1e30f; l[r] = 0.f; }

  for (int t0 = 0; t0 < SS; t0 += 32) {
    __syncthreads();   // previous tile fully consumed
    *(v8h*)(ldsK + kr * KPITCH + kc * 16)     = kreg.h2[0];
    *(v8h*)(ldsK + kr * KPITCH + kc * 16 + 8) = kreg.h2[1];
    *(v8h*)(ldsV + vr * VPITCH + vc * 16)     = vreg.h2[0];
    *(v8h*)(ldsV + vr * VPITCH + vc * 16 + 8) = vreg.h2[1];
    if (t0 + 32 < SS) {   // prefetch next tile into registers (overlaps compute)
      const _Float16* kn = kb + (size_t)(t0 + 32 + kr) * HD + kc * 16;
      kreg.h2[0] = *(const v8h*)kn;
      kreg.h2[1] = *(const v8h*)(kn + 8);
      const _Float16* vn = vb + (size_t)vr * SS + (t0 + 32) + vc * 16;
      vreg.h2[0] = *(const v8h*)vn;
      vreg.h2[1] = *(const v8h*)(vn + 8);
    }
    __syncthreads();   // tile visible to all waves

    // ---- scores: 16 rows x 32 keys (q pre-scaled by 1/8) ----
    v8f s[2];
#pragma unroll
    for (int j = 0; j < 2; ++j) {
      const _Float16* kcol = ldsK + (j * 16 + n16) * KPITCH + half * 16;
      v16h bk0 = load_b16(kcol);        // hd 0..31
      v16h bk1 = load_b16(kcol + 32);   // hd 32..63
      v8f z = {};
      z = wmma_f16(a0, bk0, z);
      s[j] = wmma_f16(a1, bk1, z);
    }
    // ---- row max over the 32 new keys (16-lane group reduction) ----
    float rmax[8];
#pragma unroll
    for (int r = 0; r < 8; ++r) rmax[r] = fmaxf(s[0][r], s[1][r]);
#pragma unroll
    for (int off = 8; off >= 1; off >>= 1)
#pragma unroll
      for (int r = 0; r < 8; ++r)
        rmax[r] = fmaxf(rmax[r], __shfl_xor(rmax[r], off, 32));
    // ---- online softmax state update ----
    float alpha[8];
#pragma unroll
    for (int r = 0; r < 8; ++r) {
      float mn = fmaxf(m[r], rmax[r]);
      alpha[r] = __expf(m[r] - mn);
      m[r] = mn;
      l[r] *= alpha[r];
    }
#pragma unroll
    for (int nt = 0; nt < 4; ++nt)
#pragma unroll
      for (int r = 0; r < 8; ++r) O[nt][r] *= alpha[r];
    // ---- probabilities + row sums ----
    v8f p[2];
    float rsum[8];
#pragma unroll
    for (int r = 0; r < 8; ++r) rsum[r] = 0.f;
#pragma unroll
    for (int j = 0; j < 2; ++j)
#pragma unroll
      for (int r = 0; r < 8; ++r) {
        float pv = __expf(s[j][r] - m[r]);
        p[j][r] = pv;
        rsum[r] += pv;
      }
#pragma unroll
    for (int off = 8; off >= 1; off >>= 1)
#pragma unroll
      for (int r = 0; r < 8; ++r) rsum[r] += __shfl_xor(rsum[r], off, 32);
#pragma unroll
    for (int r = 0; r < 8; ++r) l[r] += rsum[r];
    // ---- transpose P (C layout -> A layout) via per-wave LDS (in-order DS) ----
#pragma unroll
    for (int j = 0; j < 2; ++j)
#pragma unroll
      for (int r = 0; r < 8; ++r)
        P[(r + 8 * half) * PPITCH + j * 16 + n16] = (_Float16)p[j][r];
    v16h pa;
    {
      union { v16h v; v8h h2[2]; } u;
      const _Float16* lp = P + n16 * PPITCH;
      u.h2[0] = *(const v8h*)(lp + half * 16);
      u.h2[1] = *(const v8h*)(lp + half * 16 + 8);
      pa = u.v;
    }
    // ---- O += P(16x32) * V(32x64) ----
#pragma unroll
    for (int nt = 0; nt < 4; ++nt) {
      v16h bv = load_b16(ldsV + (nt * 16 + n16) * VPITCH + half * 16);
      O[nt] = wmma_f16(pa, bv, O[nt]);
    }
  }
  // ---- normalize + store attended as [B, S, H*hd] f32 ----
  float inv[8];
#pragma unroll
  for (int r = 0; r < 8; ++r) inv[r] = 1.0f / l[r];
#pragma unroll
  for (int nt = 0; nt < 4; ++nt) {
    int n = nt * 16 + n16;
#pragma unroll
    for (int r = 0; r < 8; ++r) {
      int row = row0 + r + 8 * half;
      att[(((size_t)b * SS + row) * NH + h) * HD + n] = O[nt][r] * inv[r];
    }
  }
}

// ---------------------------------------------------------------------------
// Kernel 5: LayerNorm over last dim (1024), one block per row
// ---------------------------------------------------------------------------
__global__ void ln_kernel(const float* __restrict__ att,
                          const float* __restrict__ gamma,
                          const float* __restrict__ beta,
                          float* __restrict__ out) {
  __shared__ float ssum[8], ssq[8];
  int row = blockIdx.x;
  int tid = threadIdx.x;          // 256 threads, 4 floats each
  const float* p = att + (size_t)row * DIMC;
  float4 x = *(const float4*)(p + tid * 4);
  float sum = x.x + x.y + x.z + x.w;
  float sq = x.x * x.x + x.y * x.y + x.z * x.z + x.w * x.w;
#pragma unroll
  for (int off = 16; off >= 1; off >>= 1) {
    sum += __shfl_xor(sum, off, 32);
    sq += __shfl_xor(sq, off, 32);
  }
  int wv = tid >> 5, lane = tid & 31;
  if (lane == 0) { ssum[wv] = sum; ssq[wv] = sq; }
  __syncthreads();
  float tsum = 0.f, tsq = 0.f;
#pragma unroll
  for (int i = 0; i < 8; ++i) { tsum += ssum[i]; tsq += ssq[i]; }
  float mean = tsum * (1.0f / DIMC);
  float var = tsq * (1.0f / DIMC) - mean * mean;
  float rstd = rsqrtf(var + 1e-5f);
  float4 g = *(const float4*)(gamma + tid * 4);
  float4 bt = *(const float4*)(beta + tid * 4);
  float4 o;
  o.x = (x.x - mean) * rstd * g.x + bt.x;
  o.y = (x.y - mean) * rstd * g.y + bt.y;
  o.z = (x.z - mean) * rstd * g.z + bt.z;
  o.w = (x.w - mean) * rstd * g.w + bt.w;
  *(float4*)(out + (size_t)row * DIMC + tid * 4) = o;
}

// ---------------------------------------------------------------------------
extern "C" void kernel_launch(void* const* d_in, const int* in_sizes, int n_in,
                              void* d_out, int out_size, void* d_ws, size_t ws_size,
                              hipStream_t stream) {
  (void)in_sizes; (void)n_in; (void)out_size; (void)ws_size;
  const float* q  = (const float*)d_in[0];
  const float* k  = (const float*)d_in[1];
  const float* v  = (const float*)d_in[2];
  const float* Wq = (const float*)d_in[3];
  const float* bq = (const float*)d_in[4];
  const float* Wk = (const float*)d_in[5];
  const float* bk = (const float*)d_in[6];
  const float* Wv = (const float*)d_in[7];
  const float* bv = (const float*)d_in[8];
  const float* gamma = (const float*)d_in[9];
  const float* beta  = (const float*)d_in[10];

  const size_t NX = (size_t)BB * SS * DIMC;   // 4,194,304 elems per activation
  const size_t NW = (size_t)NH * DIMC * HD;   // 1,048,576 elems per weight

  char* ws = (char*)d_ws;
  _Float16* X16  = (_Float16*)ws;                              // 3*NX f16
  _Float16* Wt16 = (_Float16*)(ws + 3 * NX * 2);               // 3*NW f16
  _Float16* qp   = (_Float16*)(ws + 3 * NX * 2 + 3 * NW * 2);  // NX f16
  _Float16* kp   = qp + NX;                                    // NX f16
  _Float16* vT   = kp + NX;                                    // NX f16
  float*    att  = (float*)ws;  // reuse dead X16 region (16MB <= 24MB)

  cvt_f32_f16_kernel<<<NX / 1024, 256, 0, stream>>>(q, X16);
  cvt_f32_f16_kernel<<<NX / 1024, 256, 0, stream>>>(k, X16 + NX);
  cvt_f32_f16_kernel<<<NX / 1024, 256, 0, stream>>>(v, X16 + 2 * NX);
  pack_w_kernel<<<NW / 256, 256, 0, stream>>>(Wq, Wt16);
  pack_w_kernel<<<NW / 256, 256, 0, stream>>>(Wk, Wt16 + NW);
  pack_w_kernel<<<NW / 256, 256, 0, stream>>>(Wv, Wt16 + 2 * NW);

  dim3 pg(SS / 64, NH, BB);   // 4 waves/block, 16 rows/wave
  proj_kernel<<<pg, 128, 0, stream>>>(X16,          Wt16,          bq, qp, 0);
  proj_kernel<<<pg, 128, 0, stream>>>(X16 + NX,     Wt16 + NW,     bk, kp, 1);
  proj_kernel<<<pg, 128, 0, stream>>>(X16 + 2 * NX, Wt16 + 2 * NW, bv, vT, 2);

  attn_kernel<<<pg, 128, 0, stream>>>(qp, kp, vT, att);

  ln_kernel<<<BB * SS, 256, 0, stream>>>(att, gamma, beta, (float*)d_out);
}
